// MaskEstimationModule_27453430956751
// MI455X (gfx1250) — matrix operand order repaired and checked by reading, not compiled, not run on hardware
//
#include <hip/hip_runtime.h>
#include <hip/hip_bf16.h>

// ---------------------------------------------------------------------------
// Problem constants (from the reference)
// ---------------------------------------------------------------------------
constexpr int M_ROWS   = 8 * 512;     // B*T = 4096
constexpr int NBANDS   = 62;
constexpr int D        = 256;         // hidden
constexpr int E        = 512;         // hidden*2 (GLU input)
constexpr int NOUT     = 34;          // true padded per-band output
constexpr int NOUT_PAD = 48;          // padded to 3 WMMA N-tiles
constexpr int FREQ     = 1025;
constexpr int LDS_PITCH = D + 8;      // 264 bf16: 16B-aligned rows, staggers banks

typedef __attribute__((ext_vector_type(16))) __bf16 v16bf;
typedef __attribute__((ext_vector_type(8)))  __bf16 v8bf;
typedef __attribute__((ext_vector_type(8)))  float  v8f;
typedef int v4i32 __attribute__((vector_size(16)));   // matches builtin "V4i"

#if defined(__AMDGCN__) && __has_builtin(__builtin_amdgcn_global_load_async_to_lds_b128)
#define HAVE_ASYNC_LDS 1
typedef __attribute__((address_space(1))) v4i32* as1_v4i_p;
typedef __attribute__((address_space(3))) v4i32* as3_v4i_p;
#else
#define HAVE_ASYNC_LDS 0
#endif

static __device__ __forceinline__ v8f vzero8() {
  v8f z = {0.f, 0.f, 0.f, 0.f, 0.f, 0.f, 0.f, 0.f};
  return z;
}

// Copy one 16-byte chunk global -> LDS, via async-LDS engine when available.
static __device__ __forceinline__ void copy16_to_lds(__bf16* dst_lds, const __bf16* src_g) {
#if HAVE_ASYNC_LDS
  __builtin_amdgcn_global_load_async_to_lds_b128(
      (as1_v4i_p)(src_g), (as3_v4i_p)(dst_lds), 0, 0);
#else
  *(v4i32*)dst_lds = *(const v4i32*)src_g;
#endif
}

static __device__ __forceinline__ void wait_async_lds() {
#if HAVE_ASYNC_LDS
#if __has_builtin(__builtin_amdgcn_s_wait_asynccnt)
  __builtin_amdgcn_s_wait_asynccnt(0);
#else
  asm volatile("s_wait_asynccnt 0" ::: "memory");
#endif
#endif
}

// ---------------------------------------------------------------------------
// Kernel 0: weight prep. W1 (62,256,512) f32 -> W1t (62,512,256) bf16
//           W2 (62,256,34) f32 -> W2t (62,48,256) bf16 (zero-padded cols)
// ---------------------------------------------------------------------------
__global__ void k_prep_weights(const float* __restrict__ W1,
                               const float* __restrict__ W2,
                               __bf16* __restrict__ W1t,
                               __bf16* __restrict__ W2t) {
  size_t i = (size_t)blockIdx.x * blockDim.x + threadIdx.x;
  const size_t n1 = (size_t)NBANDS * E * D;
  if (i < n1) {
    int    k = (int)(i % D);
    size_t r = i / D;
    int    e = (int)(r % E);
    int    n = (int)(r / E);
    W1t[i] = (__bf16)W1[((size_t)n * D + k) * E + e];
  }
  const size_t n2 = (size_t)NBANDS * NOUT_PAD * D;
  if (i < n2) {
    int    k = (int)(i % D);
    size_t r = i / D;
    int    e = (int)(r % NOUT_PAD);
    int    n = (int)(r / NOUT_PAD);
    W2t[i] = (e < NOUT) ? (__bf16)W2[((size_t)n * D + k) * NOUT + e] : (__bf16)0.0f;
  }
}

// ---------------------------------------------------------------------------
// Kernel 1: per-band RMSNorm, f32 -> bf16, relayout (B,T,N,D) -> (N, M, D).
// ---------------------------------------------------------------------------
__global__ __launch_bounds__(256) void k_rmsnorm(
    const float* __restrict__ x, const float* __restrict__ rw,
    __bf16* __restrict__ xn) {
  int wave = threadIdx.x >> 5;
  int lane = threadIdx.x & 31;
  size_t gid = (size_t)blockIdx.x * 8 + wave;          // [0, M_ROWS*NBANDS)
  int n = (int)(gid % NBANDS);
  int m = (int)(gid / NBANDS);

  const float* row = x + ((size_t)m * NBANDS + n) * D;
  float v[8];
  float ss = 0.f;
#pragma unroll
  for (int i = 0; i < 8; ++i) {
    v[i] = row[lane + 32 * i];
    ss += v[i] * v[i];
  }
#pragma unroll
  for (int s = 16; s >= 1; s >>= 1) ss += __shfl_xor(ss, s, 32);
  float scale = rsqrtf(ss * (1.0f / D) + 1e-6f);

  __bf16* orow = xn + ((size_t)n * M_ROWS + m) * D;
#pragma unroll
  for (int i = 0; i < 8; ++i) {
    int d = lane + 32 * i;
    orow[d] = (__bf16)(v[i] * scale * rw[n * D + d]);
  }
}

// ---------------------------------------------------------------------------
// WMMA fragment helpers, this file's own contiguous K order (not the ISA table order), same for A and B.
// A (16x32, MxK): lane%16 = row; lane-half selects contiguous K-range of 16.
// B (32x16, KxN): lane%16 = col; lane-half selects contiguous K-range of 16.
// C/D (16x16):    VGPR v, lane l -> row v + 8*(l>>4), col l&15.
// ---------------------------------------------------------------------------
static __device__ __forceinline__ v16bf load_a_frag(const __bf16* arow, int kb, int half) {
  v8bf a0 = *(const v8bf*)(arow + kb + 16 * half);
  v8bf a1 = *(const v8bf*)(arow + kb + 16 * half + 8);
  v16bf f;
#pragma unroll
  for (int t = 0; t < 8; ++t) { f[t] = a0[t]; f[t + 8] = a1[t]; }
  return f;
}

static __device__ __forceinline__ v16bf load_b_frag(const __bf16* p) {
  // 16 contiguous bf16 starting at p (two 16B LDS loads)
  v8bf b0 = *(const v8bf*)(p);
  v8bf b1 = *(const v8bf*)(p + 8);
  v16bf f;
#pragma unroll
  for (int t = 0; t < 8; ++t) { f[t] = b0[t]; f[t + 8] = b1[t]; }
  return f;
}

// ---------------------------------------------------------------------------
// Kernel 2: grouped GEMM1 (per band: 4096x256 @ 256x512) + bias + GLU.
// Grid (M/128, 4 col-groups, 62 bands), 8 waves/block.
// Block stages its 128 weight rows (4 a-tiles + 4 GLU-partner tiles, 64KB)
// into LDS via the async global->LDS engine, then each wave runs
// 64 v_wmma_f32_16x16x32_bf16 with B fragments from LDS.
// ---------------------------------------------------------------------------
__global__ __launch_bounds__(256, 1) void k_gemm1_glu(
    const __bf16* __restrict__ xn, const __bf16* __restrict__ W1t,
    const float* __restrict__ b1, __bf16* __restrict__ H) {
  __shared__ __bf16 smem[128 * LDS_PITCH];   // 67,584 B

  int n = blockIdx.z;
  int g = blockIdx.y;                        // column group: tiles j = 4g..4g+3
  int wave = threadIdx.x >> 5;
  int lane = threadIdx.x & 31;
  int lane15 = lane & 15;
  int half = lane >> 4;
  int m0 = blockIdx.x * 128 + wave * 16;

  const __bf16* wbase = W1t + (size_t)n * E * D;

  // ---- stage weights: smem rows 0..63 = a-cols [64g,64g+64),
  //                     smem rows 64..127 = g-cols [256+64g, ...)
  {
    const int NCHUNK = 128 * 32;             // 16B chunks
    for (int c = threadIdx.x; c < NCHUNK; c += 256) {
      int row = c >> 5;
      int cc  = c & 31;
      int srcrow = (row < 64) ? (64 * g + row) : (256 + 64 * g + (row - 64));
      copy16_to_lds(&smem[row * LDS_PITCH + cc * 8],
                    wbase + (size_t)srcrow * D + cc * 8);
    }
    wait_async_lds();
    __syncthreads();
  }

  const __bf16* arow = xn + ((size_t)n * M_ROWS + m0 + lane15) * D;

  v8f acc_a[4], acc_g[4];
#pragma unroll
  for (int i = 0; i < 4; ++i) { acc_a[i] = vzero8(); acc_g[i] = vzero8(); }

#pragma unroll
  for (int kb = 0; kb < D; kb += 32) {
    if (kb + 32 < D) {
      __builtin_prefetch(arow + kb + 32 + 8 * half, 0, 3);
    }
    v16bf afrag = load_a_frag(arow, kb, half);
#pragma unroll
    for (int i = 0; i < 4; ++i) {
      v16bf bfa = load_b_frag(&smem[(i * 16 + lane15) * LDS_PITCH + kb + 16 * half]);
      v16bf bfg = load_b_frag(&smem[(64 + i * 16 + lane15) * LDS_PITCH + kb + 16 * half]);
      acc_a[i] = __builtin_amdgcn_wmma_f32_16x16x32_bf16(
          false, afrag, false, bfa, (short)0, acc_a[i], false, false);
      acc_g[i] = __builtin_amdgcn_wmma_f32_16x16x32_bf16(
          false, afrag, false, bfg, (short)0, acc_g[i], false, false);
    }
  }

  // Epilogue: bias + GLU (a * sigmoid(g)), write bf16 H in (N, M, D) layout.
#pragma unroll
  for (int i = 0; i < 4; ++i) {
    int j  = g * 4 + i;
    int ca = j * 16 + lane15;
    float biasa = b1[n * E + ca];
    float biasg = b1[n * E + 256 + ca];
#pragma unroll
    for (int v = 0; v < 8; ++v) {
      int r = m0 + v + 8 * half;
      float a = acc_a[i][v] + biasa;
      float gg = acc_g[i][v] + biasg;
      float hv = a * (1.0f / (1.0f + __expf(-gg)));
      H[((size_t)n * M_ROWS + r) * D + ca] = (__bf16)hv;
    }
  }
}

// ---------------------------------------------------------------------------
// Kernel 3: grouped GEMM2 (per band: 4096x256 @ 256x48pad) + bias + tanh.
// Whole 48-row band slab (24KB) staged into LDS; 8 waves x 3 N-tiles each.
// Writes f32 o to scratch in (M, NBANDS, 48) layout.
// ---------------------------------------------------------------------------
__global__ __launch_bounds__(256, 1) void k_gemm2_tanh(
    const __bf16* __restrict__ H, const __bf16* __restrict__ W2t,
    const float* __restrict__ b2, float* __restrict__ ows) {
  __shared__ __bf16 smem[NOUT_PAD * LDS_PITCH];   // 25,344 B

  int n = blockIdx.y;
  int wave = threadIdx.x >> 5;
  int lane = threadIdx.x & 31;
  int lane15 = lane & 15;
  int half = lane >> 4;
  int m0 = (blockIdx.x * 8 + wave) * 16;

  const __bf16* wbase = W2t + (size_t)n * NOUT_PAD * D;

  {
    const int NCHUNK = NOUT_PAD * 32;        // 1536 x 16B
    for (int c = threadIdx.x; c < NCHUNK; c += 256) {
      int row = c >> 5;
      int cc  = c & 31;
      copy16_to_lds(&smem[row * LDS_PITCH + cc * 8],
                    wbase + (size_t)row * D + cc * 8);
    }
    wait_async_lds();
    __syncthreads();
  }

  const __bf16* arow = H + ((size_t)n * M_ROWS + m0 + lane15) * D;

  v8f acc[3];
#pragma unroll
  for (int i = 0; i < 3; ++i) acc[i] = vzero8();

#pragma unroll
  for (int kb = 0; kb < D; kb += 32) {
    if (kb + 32 < D) {
      __builtin_prefetch(arow + kb + 32 + 8 * half, 0, 3);
    }
    v16bf afrag = load_a_frag(arow, kb, half);
#pragma unroll
    for (int nt = 0; nt < 3; ++nt) {
      v16bf bfr = load_b_frag(&smem[(nt * 16 + lane15) * LDS_PITCH + kb + 16 * half]);
      acc[nt] = __builtin_amdgcn_wmma_f32_16x16x32_bf16(
          false, afrag, false, bfr, (short)0, acc[nt], false, false);
    }
  }

#pragma unroll
  for (int nt = 0; nt < 3; ++nt) {
    int col = nt * 16 + lane15;
    float bias = (col < NOUT) ? b2[n * NOUT + col] : 0.0f;
#pragma unroll
    for (int v = 0; v < 8; ++v) {
      int r = m0 + v + 8 * half;
      ows[((size_t)r * NBANDS + n) * NOUT_PAD + col] = tanhf(acc[nt][v] + bias);
    }
  }
}

// ---------------------------------------------------------------------------
// Kernel 4: gather per-band mask columns and apply to original_spec.
// band(f) = floor((62f+61)/1025); pos = 2*(f - floor(band*1025/62)) + c.
// ---------------------------------------------------------------------------
__global__ __launch_bounds__(256) void k_apply_mask(
    const float* __restrict__ spec, const float* __restrict__ ows,
    float* __restrict__ out, int total) {
  int idx = blockIdx.x * 256 + threadIdx.x;
  if (idx >= total) return;
  int c = idx & 1;
  int f = (idx >> 1) % FREQ;
  int m = idx / (FREQ * 2);
  int band = (62 * f + 61) / 1025;
  int base = (band * 1025) / 62;
  int pos  = 2 * (f - base) + c;
  out[idx] = spec[idx] * ows[((size_t)m * NBANDS + band) * NOUT_PAD + pos];
}

// ---------------------------------------------------------------------------
// Launch
// ---------------------------------------------------------------------------
extern "C" void kernel_launch(void* const* d_in, const int* in_sizes, int n_in,
                              void* d_out, int out_size, void* d_ws, size_t ws_size,
                              hipStream_t stream) {
  const float* band_features = (const float*)d_in[0];  // (8,512,62,256)
  const float* original_spec = (const float*)d_in[1];  // (8,512,1025,2)
  const float* rms_w         = (const float*)d_in[2];  // (62,256)
  const float* W1            = (const float*)d_in[3];  // (62,256,512)
  const float* b1            = (const float*)d_in[4];  // (62,512)
  const float* W2            = (const float*)d_in[5];  // (62,256,34)
  const float* b2            = (const float*)d_in[6];  // (62,34)
  float* out = (float*)d_out;                          // (8,512,1025,2)

  const size_t w1t_bytes = (size_t)NBANDS * E * D * sizeof(__bf16);          // 16,252,928
  const size_t w2t_bytes = (size_t)NBANDS * NOUT_PAD * D * sizeof(__bf16);   //  1,523,712
  const size_t xn_bytes  = (size_t)NBANDS * M_ROWS * D * sizeof(__bf16);     // 130,023,424

  char* ws = (char*)d_ws;
  __bf16* W1t = (__bf16*)(ws);
  __bf16* W2t = (__bf16*)(ws + w1t_bytes);
  __bf16* xn  = (__bf16*)(ws + w1t_bytes + w2t_bytes);
  __bf16* H   = (__bf16*)(ws + w1t_bytes + w2t_bytes + xn_bytes);
  // ows reuses the xn region (xn dead after k_gemm1_glu; stream-ordered).
  float*  ows = (float*)(ws + w1t_bytes + w2t_bytes);

  {
    int blocks = (int)(((size_t)NBANDS * E * D + 255) / 256);
    k_prep_weights<<<blocks, 256, 0, stream>>>(W1, W2, W1t, W2t);
  }
  {
    int rows = M_ROWS * NBANDS;            // 253952
    k_rmsnorm<<<rows / 8, 256, 0, stream>>>(band_features, rms_w, xn);
  }
  {
    dim3 grid(M_ROWS / 128, 4, NBANDS);    // (32, 4, 62)
    k_gemm1_glu<<<grid, 256, 0, stream>>>(xn, W1t, b1, H);
  }
  {
    dim3 grid(M_ROWS / 128, NBANDS);       // (32, 62)
    k_gemm2_tanh<<<grid, 256, 0, stream>>>(H, W2t, b2, ows);
  }
  {
    int total = M_ROWS * FREQ * 2;         // 8,396,800 == out_size
    k_apply_mask<<<(total + 255) / 256, 256, 0, stream>>>(original_spec, ows, out, total);
  }
}
